// MoonVitEncoderLayer_87333864996970
// MI455X (gfx1250) — matrix operand order repaired and checked by reading, not compiled, not run on hardware
//
#include <hip/hip_runtime.h>
#include <hip/hip_bf16.h>

typedef unsigned short ushort_t;
typedef unsigned int uint_t;

#define LL 4096
#define DD 1024
#define HH 16
#define HDIM 64
#define MM 4096

typedef __attribute__((ext_vector_type(16))) __bf16 bf16x16;
typedef __attribute__((ext_vector_type(8)))  float  floatx8;
typedef __attribute__((ext_vector_type(4))) unsigned int uint32x4;
typedef __attribute__((ext_vector_type(8))) int int32x8;
typedef __attribute__((ext_vector_type(4))) int int32x4;

union FragAB {
    uint4   u[2];
    bf16x16 v;
};

__device__ __forceinline__ ushort_t f2bf(float f) {
    uint_t u = __float_as_uint(f);
    uint_t r = (u + 0x7FFFu + ((u >> 16) & 1u)) >> 16;
    return (ushort_t)r;
}

__device__ __forceinline__ void zero8(floatx8& v) {
#pragma unroll
    for (int i = 0; i < 8; ++i) v[i] = 0.0f;
}

// ---- v_permlane16-based XOR butterflies within each 16-lane group ---------
__device__ __forceinline__ float pl16(float v, unsigned s0, unsigned s1) {
    int r = __builtin_amdgcn_permlane16(__float_as_int(v), __float_as_int(v),
                                        (int)s0, (int)s1, false, false);
    return __int_as_float(r);
}
__device__ __forceinline__ float bfly_max16(float v) {
    v = fmaxf(v, pl16(v, 0x67452301u, 0xEFCDAB89u)); // xor 1
    v = fmaxf(v, pl16(v, 0x54761032u, 0xDCFE98BAu)); // xor 2
    v = fmaxf(v, pl16(v, 0x32107654u, 0xBA98FEDCu)); // xor 4
    v = fmaxf(v, pl16(v, 0xFEDCBA98u, 0x76543210u)); // xor 8
    return v;
}
__device__ __forceinline__ float bfly_sum16(float v) {
    v += pl16(v, 0x67452301u, 0xEFCDAB89u);
    v += pl16(v, 0x54761032u, 0xDCFE98BAu);
    v += pl16(v, 0x32107654u, 0xBA98FEDCu);
    v += pl16(v, 0xFEDCBA98u, 0x76543210u);
    return v;
}

// ---------------------------------------------------------------------------
// Tensor Data Mover: 2D tile (tile_dim0=32 elems of 2B, tile_dim1=128 rows)
// from global (row stride = strideElems) into LDS at lds_addr.
// ---------------------------------------------------------------------------
__device__ __forceinline__ void tdm_load_tile_128x32(
    unsigned lds_addr, const ushort_t* gptr, unsigned strideElems) {
    unsigned long long ga = (unsigned long long)(uintptr_t)gptr;
    uint32x4 g0;
    g0[0] = 1u;                                   // count=1, user mode
    g0[1] = lds_addr;                             // LDS byte address
    g0[2] = (unsigned)ga;                         // global addr [31:0]
    g0[3] = ((unsigned)(ga >> 32) & 0x01FFFFFFu)  // global addr [56:32]
            | 0x80000000u;                        // type=2 ("image")
    const unsigned tdim = 1u << 20;               // generous tensor dims
    int32x8 g1;
    g1[0] = (int)0x00010000u;                     // mask=0, data_size=1 (2B)
    g1[1] = (int)((tdim & 0xFFFFu) << 16);        // tensor_dim0[15:0]
    g1[2] = (int)((tdim >> 16) | ((tdim & 0xFFFFu) << 16)); // d0 hi | d1 lo
    g1[3] = (int)((tdim >> 16) | (32u << 16));    // d1 hi | tile_dim0=32
    g1[4] = (int)128u;                            // tile_dim1=128, tile_dim2=0
    g1[5] = (int)strideElems;                     // tensor_dim0_stride lo
    g1[6] = 0;                                    // stride0 hi | stride1 lo
    g1[7] = 0;                                    // stride1 hi
    int32x4 gz = {0, 0, 0, 0};
#if defined(__clang_major__) && (__clang_major__ >= 23)
    int32x8 gz8 = {0, 0, 0, 0, 0, 0, 0, 0};
    __builtin_amdgcn_tensor_load_to_lds(g0, g1, gz, gz, gz8, 0);
#else
    __builtin_amdgcn_tensor_load_to_lds(g0, g1, gz, gz, 0);
#endif
}

// ---------------------------------------------------------------------------
// Weight cast + transpose: w[k][n] fp32 -> wT[n][k] bf16
// ---------------------------------------------------------------------------
__global__ __launch_bounds__(256) void cast_transpose_kernel(
    const float* __restrict__ w, ushort_t* __restrict__ wT, int K, int N) {
    size_t idx = (size_t)blockIdx.x * blockDim.x + threadIdx.x;
    size_t total = (size_t)K * N;
    if (idx >= total) return;
    int n = (int)(idx / K);
    int k = (int)(idx % K);
    wT[idx] = f2bf(w[(size_t)k * N + n]);
}

// ---------------------------------------------------------------------------
// LayerNorm (row of 1024) fp32 in -> bf16 out
// ---------------------------------------------------------------------------
__global__ __launch_bounds__(256) void layernorm_kernel(
    const float* __restrict__ x, const float* __restrict__ g,
    const float* __restrict__ b, ushort_t* __restrict__ out) {
    __shared__ float s1[256];
    __shared__ float s2[256];
    int row = blockIdx.x;
    int t = threadIdx.x;
    const float4 v = *(const float4*)(x + (size_t)row * DD + t * 4);
    float sum = v.x + v.y + v.z + v.w;
    float sq  = v.x * v.x + v.y * v.y + v.z * v.z + v.w * v.w;
    s1[t] = sum; s2[t] = sq;
    __syncthreads();
#pragma unroll
    for (int off = 128; off > 0; off >>= 1) {
        if (t < off) { s1[t] += s1[t + off]; s2[t] += s2[t + off]; }
        __syncthreads();
    }
    float mu = s1[0] * (1.0f / DD);
    float var = s2[0] * (1.0f / DD) - mu * mu;
    float rstd = rsqrtf(var + 1e-5f);
    int c = t * 4;
    float o0 = (v.x - mu) * rstd * g[c + 0] + b[c + 0];
    float o1 = (v.y - mu) * rstd * g[c + 1] + b[c + 1];
    float o2 = (v.z - mu) * rstd * g[c + 2] + b[c + 2];
    float o3 = (v.w - mu) * rstd * g[c + 3] + b[c + 3];
    ushort_t* dst = out + (size_t)row * DD + c;
    dst[0] = f2bf(o0); dst[1] = f2bf(o1); dst[2] = f2bf(o2); dst[3] = f2bf(o3);
}

// ---------------------------------------------------------------------------
// WMMA GEMM with TDM double-buffered staging.
// C[M][N] = A[M][K](bf16,row-major) @ BT[N][K](bf16)^T
// Block tile 128x128, BK=32, 8 waves (2x4) of 64x32 each.
// Wave 0 drives the Tensor Data Mover; tile i+1 DMA overlaps tile i compute.
// EP: 0 = C fp32 ; 1 = C+resid fp32 ; 2 = gelu(C+bias) bf16 ; 3 = C+bias+resid
// ---------------------------------------------------------------------------
template <int EP>
__global__ __launch_bounds__(256) void gemm_bf16_kernel(
    const ushort_t* __restrict__ A, const ushort_t* __restrict__ BT,
    float* __restrict__ Cf, ushort_t* __restrict__ Cb,
    const float* __restrict__ bias, const float* __restrict__ resid,
    int M, int N, int K) {
    __shared__ ushort_t As[2][128 * 32];
    __shared__ ushort_t Bs[2][128 * 32];

    const int t    = threadIdx.x;
    const int wave = t >> 5;
    const int lane = t & 31;
    const int hi   = lane >> 4;
    const int ln   = lane & 15;
    const int wm   = wave >> 2;   // 0..1
    const int wn   = wave & 3;    // 0..3
    const int m0   = blockIdx.y * 128;
    const int n0   = blockIdx.x * 128;

    const unsigned ldsA = (unsigned)(uintptr_t)(&As[0][0]);
    const unsigned ldsB = (unsigned)(uintptr_t)(&Bs[0][0]);

    floatx8 acc[4][2];
#pragma unroll
    for (int mi = 0; mi < 4; ++mi)
#pragma unroll
        for (int ni = 0; ni < 2; ++ni) zero8(acc[mi][ni]);

    const ushort_t* Ag = A + (size_t)m0 * K;
    const ushort_t* Bg = BT + (size_t)n0 * K;
    const int nIt = K >> 5;

    // prologue: DMA tile 0 into buffer 0
    if (wave == 0) {
        tdm_load_tile_128x32(ldsA, Ag, (unsigned)K);
        tdm_load_tile_128x32(ldsB, Bg, (unsigned)K);
    }

    int cur = 0;
    for (int it = 0; it < nIt; ++it) {
        if (wave == 0) __builtin_amdgcn_s_wait_tensorcnt(0);
        __syncthreads();   // tile `it` visible to all; buf (1-cur) free
        if (wave == 0 && (it + 1) < nIt) {
            int k1 = (it + 1) << 5;
            tdm_load_tile_128x32(ldsA + (1 - cur) * 8192, Ag + k1, (unsigned)K);
            tdm_load_tile_128x32(ldsB + (1 - cur) * 8192, Bg + k1, (unsigned)K);
        }

        const ushort_t* Asc = &As[cur][0];
        const ushort_t* Bsc = &Bs[cur][0];
        FragAB af[4], bf[2];
#pragma unroll
        for (int mi = 0; mi < 4; ++mi) {
            int base = (wm * 64 + mi * 16 + ln) * 32;
            af[mi].u[0] = *(const uint4*)(Asc + base + hi * 16);
            af[mi].u[1] = *(const uint4*)(Asc + base + hi * 16 + 8);
        }
#pragma unroll
        for (int ni = 0; ni < 2; ++ni) {
            int base = (wn * 32 + ni * 16 + ln) * 32;
            bf[ni].u[0] = *(const uint4*)(Bsc + base + hi * 16);
            bf[ni].u[1] = *(const uint4*)(Bsc + base + hi * 16 + 8);
        }
#pragma unroll
        for (int mi = 0; mi < 4; ++mi)
#pragma unroll
            for (int ni = 0; ni < 2; ++ni)
                acc[mi][ni] = __builtin_amdgcn_wmma_f32_16x16x32_bf16(
                    false, af[mi].v, false, bf[ni].v, (short)0, acc[mi][ni],
                    false, false);
        __syncthreads();   // compute on `cur` done before it is re-DMA'd
        cur ^= 1;
    }

#pragma unroll
    for (int mi = 0; mi < 4; ++mi) {
#pragma unroll
        for (int ni = 0; ni < 2; ++ni) {
#pragma unroll
            for (int rr = 0; rr < 8; ++rr) {
                int row = m0 + wm * 64 + mi * 16 + rr + 8 * hi;
                int col = n0 + wn * 32 + ni * 16 + ln;
                float v = acc[mi][ni][rr];
                size_t off = (size_t)row * N + col;
                if (EP == 0) {
                    Cf[off] = v;
                } else if (EP == 1) {
                    Cf[off] = v + resid[off];
                } else if (EP == 2) {
                    v += bias[col];
                    v = 0.5f * v * (1.0f + erff(v * 0.70710678118654752f));
                    Cb[off] = f2bf(v);
                } else {
                    Cf[off] = v + bias[col] + resid[off];
                }
            }
        }
    }
}

// ---------------------------------------------------------------------------
// RoPE + pack: qkv fp32 [L][3*D] -> qp/kp bf16 [H][L][HD], vt bf16 [H][HD][L]
// ---------------------------------------------------------------------------
__global__ __launch_bounds__(256) void rope_pack_kernel(
    const float* __restrict__ qkv, const float* __restrict__ fc,
    ushort_t* __restrict__ qp, ushort_t* __restrict__ kp,
    ushort_t* __restrict__ vt) {
    size_t idx = (size_t)blockIdx.x * blockDim.x + threadIdx.x;
    if (idx >= (size_t)LL * HH * (HDIM / 2)) return;
    int l   = (int)(idx / (HH * (HDIM / 2)));
    int rem = (int)(idx % (HH * (HDIM / 2)));
    int h = rem / (HDIM / 2);
    int i = rem % (HDIM / 2);

    size_t base = (size_t)l * (3 * DD) + h * HDIM + 2 * i;
    float qa = qkv[base],            qb = qkv[base + 1];
    float ka = qkv[base + DD],       kb = qkv[base + DD + 1];
    float va = qkv[base + 2 * DD],   vb = qkv[base + 2 * DD + 1];
    float c = fc[((size_t)l * (HDIM / 2) + i) * 2 + 0];
    float s = fc[((size_t)l * (HDIM / 2) + i) * 2 + 1];

    size_t po = ((size_t)h * LL + l) * HDIM + 2 * i;
    qp[po]     = f2bf(qa * c - qb * s);
    qp[po + 1] = f2bf(qa * s + qb * c);
    kp[po]     = f2bf(ka * c - kb * s);
    kp[po + 1] = f2bf(ka * s + kb * c);
    vt[((size_t)h * HDIM + 2 * i) * LL + l]     = f2bf(va);
    vt[((size_t)h * HDIM + 2 * i + 1) * LL + l] = f2bf(vb);
}

// ---------------------------------------------------------------------------
// Flash attention, block-diagonal segments. One block = (head, 128 q rows),
// 8 waves of 16 q rows each; kv tiles of 32.
// ---------------------------------------------------------------------------
__global__ __launch_bounds__(256) void attn_kernel(
    const ushort_t* __restrict__ qp, const ushort_t* __restrict__ kp,
    const ushort_t* __restrict__ vt, const int* __restrict__ cu,
    ushort_t* __restrict__ ao) {
    __shared__ ushort_t plds[8 * 16 * 32];

    const int b  = blockIdx.x;
    const int h  = b >> 5;          // 16 heads
    const int qb = b & 31;          // 32 q-blocks
    const int q0 = qb * 128;

    int cs0 = cu[0], cs1 = cu[1], cs2 = cu[2], cs3 = cu[3], cs4 = cu[4];
    int kvs, kve;
    if (q0 < cs1)      { kvs = cs0; kve = cs1; }
    else if (q0 < cs2) { kvs = cs1; kve = cs2; }
    else if (q0 < cs3) { kvs = cs2; kve = cs3; }
    else               { kvs = cs3; kve = cs4; }

    const int wave = threadIdx.x >> 5;
    const int lane = threadIdx.x & 31;
    const int hi   = lane >> 4;
    const int ln   = lane & 15;
    const int qr0  = q0 + wave * 16;
    const float scale = 0.125f;   // 1/sqrt(64)

    // Q fragments (A layout), loaded once: k-steps hd 0..31, 32..63
    FragAB aq[2];
#pragma unroll
    for (int kk = 0; kk < 2; ++kk) {
        size_t addr = ((size_t)h * LL + (qr0 + ln)) * HDIM + kk * 32 + hi * 16;
        aq[kk].u[0] = *(const uint4*)(qp + addr);
        aq[kk].u[1] = *(const uint4*)(qp + addr + 8);
    }

    float m[8], lsum[8];
    floatx8 o[4];
#pragma unroll
    for (int rr = 0; rr < 8; ++rr) { m[rr] = -1e30f; lsum[rr] = 0.0f; }
#pragma unroll
    for (int n = 0; n < 4; ++n) zero8(o[n]);

    ushort_t* pw = plds + wave * 512;

    for (int kv0 = kvs; kv0 + 32 <= kve; kv0 += 32) {
        // ---- S = Q @ K^T for 16(q) x 32(kv) tile ----
        floatx8 s[2];
        zero8(s[0]); zero8(s[1]);
#pragma unroll
        for (int j = 0; j < 2; ++j) {
            FragAB bk[2];
#pragma unroll
            for (int kk = 0; kk < 2; ++kk) {
                size_t addr = ((size_t)h * LL + (kv0 + j * 16 + ln)) * HDIM +
                              kk * 32 + hi * 16;
                bk[kk].u[0] = *(const uint4*)(kp + addr);
                bk[kk].u[1] = *(const uint4*)(kp + addr + 8);
            }
            s[j] = __builtin_amdgcn_wmma_f32_16x16x32_bf16(
                false, aq[0].v, false, bk[0].v, (short)0, s[j], false, false);
            s[j] = __builtin_amdgcn_wmma_f32_16x16x32_bf16(
                false, aq[1].v, false, bk[1].v, (short)0, s[j], false, false);
        }

        // ---- online softmax (permlane16 butterflies, no LDS traffic) ----
        float p0[8], p1[8];
#pragma unroll
        for (int rr = 0; rr < 8; ++rr) {
            float v0 = s[0][rr] * scale;
            float v1 = s[1][rr] * scale;
            float mx = bfly_max16(fmaxf(v0, v1));
            float mn = fmaxf(m[rr], mx);
            float al = __expf(m[rr] - mn);
            float e0 = __expf(v0 - mn);
            float e1 = __expf(v1 - mn);
            float rs = bfly_sum16(e0 + e1);
            lsum[rr] = lsum[rr] * al + rs;
            m[rr] = mn;
#pragma unroll
            for (int n = 0; n < 4; ++n) o[n][rr] *= al;
            p0[rr] = e0; p1[rr] = e1;
        }

        // ---- P to LDS (accumulator layout -> A-matrix layout) ----
#pragma unroll
        for (int rr = 0; rr < 8; ++rr) {
            pw[(rr + 8 * hi) * 32 + ln]      = f2bf(p0[rr]);
            pw[(rr + 8 * hi) * 32 + 16 + ln] = f2bf(p1[rr]);
        }
        asm volatile("s_wait_dscnt 0" ::: "memory");
        FragAB ap;
        ap.u[0] = *(const uint4*)(pw + ln * 32 + hi * 16);
        ap.u[1] = *(const uint4*)(pw + ln * 32 + hi * 16 + 8);

        // ---- O += P @ V ----
#pragma unroll
        for (int n = 0; n < 4; ++n) {
            FragAB bv;
            size_t addr = ((size_t)h * HDIM + n * 16 + ln) * LL + kv0 + hi * 16;
            bv.u[0] = *(const uint4*)(vt + addr);
            bv.u[1] = *(const uint4*)(vt + addr + 8);
            o[n] = __builtin_amdgcn_wmma_f32_16x16x32_bf16(
                false, ap.v, false, bv.v, (short)0, o[n], false, false);
        }
    }

    // ---- normalize + store bf16 into [L][D] (col offset h*64) ----
#pragma unroll
    for (int rr = 0; rr < 8; ++rr) {
        float inv = (lsum[rr] > 0.0f) ? (1.0f / lsum[rr]) : 0.0f;
        int row = qr0 + rr + 8 * hi;
#pragma unroll
        for (int n = 0; n < 4; ++n)
            ao[(size_t)row * DD + h * HDIM + n * 16 + ln] = f2bf(o[n][rr] * inv);
    }
}

// ---------------------------------------------------------------------------
// Host side
// ---------------------------------------------------------------------------
extern "C" void kernel_launch(void* const* d_in, const int* in_sizes, int n_in,
                              void* d_out, int out_size, void* d_ws,
                              size_t ws_size, hipStream_t stream) {
    const float* hidden = (const float*)d_in[0];
    const int*   cu     = (const int*)d_in[1];
    const float* rope   = (const float*)d_in[2];
    const float* ln0_g  = (const float*)d_in[3];
    const float* ln0_b  = (const float*)d_in[4];
    const float* ln1_g  = (const float*)d_in[5];
    const float* ln1_b  = (const float*)d_in[6];
    const float* w_qkv  = (const float*)d_in[7];
    const float* w_o    = (const float*)d_in[8];
    const float* w_fc0  = (const float*)d_in[9];
    const float* b_fc0  = (const float*)d_in[10];
    const float* w_fc1  = (const float*)d_in[11];
    const float* b_fc1  = (const float*)d_in[12];
    float* out = (float*)d_out;

    char* ws = (char*)d_ws;
    const size_t OFF_WQKVT = 0;                         // 3072*1024 bf16 = 6 MiB
    const size_t OFF_WOT   = OFF_WQKVT + 6291456;       // 1024*1024 bf16 = 2 MiB
    const size_t OFF_WFC0T = OFF_WOT + 2097152;         // 4096*1024 bf16 = 8 MiB
    const size_t OFF_WFC1T = OFF_WFC0T + 8388608;       // 1024*4096 bf16 = 8 MiB
    const size_t OFF_X     = OFF_WFC1T + 8388608;       // 4096*1024 bf16 = 8 MiB
    const size_t OFF_QKV   = OFF_X + 8388608;           // 4096*3072 f32 = 48 MiB
    const size_t OFF_G     = OFF_QKV;                   // reuse (qkv dead by then)
    const size_t OFF_QP    = OFF_QKV + 50331648;        // 8 MiB
    const size_t OFF_KP    = OFF_QP + 8388608;          // 8 MiB
    const size_t OFF_VT    = OFF_KP + 8388608;          // 8 MiB
    const size_t OFF_AO    = OFF_VT + 8388608;          // 8 MiB
    const size_t OFF_H     = OFF_AO + 8388608;          // 4096*1024 f32 = 16 MiB
    const size_t OFF_YLN   = OFF_H + 16777216;          // 8 MiB

    ushort_t* wqkvT = (ushort_t*)(ws + OFF_WQKVT);
    ushort_t* woT   = (ushort_t*)(ws + OFF_WOT);
    ushort_t* wfc0T = (ushort_t*)(ws + OFF_WFC0T);
    ushort_t* wfc1T = (ushort_t*)(ws + OFF_WFC1T);
    ushort_t* xbf   = (ushort_t*)(ws + OFF_X);
    float*    qkv   = (float*)(ws + OFF_QKV);
    ushort_t* gbf   = (ushort_t*)(ws + OFF_G);
    ushort_t* qpb   = (ushort_t*)(ws + OFF_QP);
    ushort_t* kpb   = (ushort_t*)(ws + OFF_KP);
    ushort_t* vtb   = (ushort_t*)(ws + OFF_VT);
    ushort_t* aob   = (ushort_t*)(ws + OFF_AO);
    float*    hbuf  = (float*)(ws + OFF_H);
    ushort_t* ylnb  = (ushort_t*)(ws + OFF_YLN);

    // 1) Weights -> bf16, transposed to [N][K]
    cast_transpose_kernel<<<(DD * 3 * DD + 255) / 256, 256, 0, stream>>>(
        w_qkv, wqkvT, DD, 3 * DD);
    cast_transpose_kernel<<<(DD * DD + 255) / 256, 256, 0, stream>>>(
        w_o, woT, DD, DD);
    cast_transpose_kernel<<<(DD * MM + 255) / 256, 256, 0, stream>>>(
        w_fc0, wfc0T, DD, MM);
    cast_transpose_kernel<<<(MM * DD + 255) / 256, 256, 0, stream>>>(
        w_fc1, wfc1T, MM, DD);

    // 2) LN0 -> bf16
    layernorm_kernel<<<LL, 256, 0, stream>>>(hidden, ln0_g, ln0_b, xbf);

    // 3) QKV = x @ w_qkv  (fp32 out)
    gemm_bf16_kernel<0><<<dim3(3 * DD / 128, LL / 128), 256, 0, stream>>>(
        xbf, wqkvT, qkv, nullptr, nullptr, nullptr, LL, 3 * DD, DD);

    // 4) RoPE + pack q/k/v to bf16 attention layouts
    rope_pack_kernel<<<(LL * HH * (HDIM / 2) + 255) / 256, 256, 0, stream>>>(
        qkv, rope, qpb, kpb, vtb);

    // 5) Flash attention over block-diagonal segments
    attn_kernel<<<HH * (LL / 128), 256, 0, stream>>>(qpb, kpb, vtb, cu, aob);

    // 6) h = attn @ w_o + residual (fp32)
    gemm_bf16_kernel<1><<<dim3(DD / 128, LL / 128), 256, 0, stream>>>(
        aob, woT, hbuf, nullptr, nullptr, hidden, LL, DD, DD);

    // 7) LN1 -> bf16
    layernorm_kernel<<<LL, 256, 0, stream>>>(hbuf, ln1_g, ln1_b, ylnb);

    // 8) g = gelu(y @ w_fc0 + b_fc0) -> bf16
    gemm_bf16_kernel<2><<<dim3(MM / 128, LL / 128), 256, 0, stream>>>(
        ylnb, wfc0T, nullptr, gbf, b_fc0, nullptr, LL, MM, DD);

    // 9) out = g @ w_fc1 + b_fc1 + h (fp32 to d_out)
    gemm_bf16_kernel<3><<<dim3(DD / 128, LL / 128), 256, 0, stream>>>(
        gbf, wfc1T, out, nullptr, b_fc1, hbuf, LL, DD, MM);
}
